// Attention_9981503996487
// MI455X (gfx1250) — compile-verified, matrix operand order checked by reading, not run on hardware
//
#include <hip/hip_runtime.h>
#include <hip/hip_bf16.h>

typedef __attribute__((ext_vector_type(16))) _Float16 v16h;
typedef __attribute__((ext_vector_type(8)))  _Float16 v8h;
typedef __attribute__((ext_vector_type(8)))  float    v8f;
typedef __attribute__((ext_vector_type(4)))  unsigned int u32x4;
typedef __attribute__((ext_vector_type(8)))  int       i32x8;
typedef __attribute__((ext_vector_type(4)))  int       i32x4;

#define S_LEN 4096
#define HDIM  2048
#define NHQ   16
#define NKV   4
#define DHEAD 128
#define KVDIM (NKV * DHEAD)   // 512

// ---------------------------------------------------------------------------
// Per-lane WMMA fragment loader (16x32 f16 A-fragment / transposed B-fragment).
// ISA layout: lanes 0-15 -> row = base + lane, K halves {kk..kk+7, kk+16..kk+23};
//             lanes 16-31 -> same rows, K halves {kk+8..kk+15, kk+24..kk+31}.
// ---------------------------------------------------------------------------
__device__ __forceinline__ v16h load_frag16(const _Float16* base, int ld, int row, int kk) {
  const int lane = threadIdx.x & 31;
  const _Float16* p = base + (size_t)(row + (lane & 15)) * ld + kk + ((lane >> 4) << 3);
  v16h out;
  ((v8h*)&out)[0] = *(const v8h*)(p);
  ((v8h*)&out)[1] = *(const v8h*)(p + 16);
  return out;
}

__device__ __forceinline__ v8f wmma_f16(v16h a, v16h b, v8f c) {
  return __builtin_amdgcn_wmma_f32_16x16x32_f16(false, a, false, b, (short)0, c, false, false);
}

__device__ __forceinline__ float half16_max(float v) {
  v = fmaxf(v, __shfl_xor(v, 1, 32));
  v = fmaxf(v, __shfl_xor(v, 2, 32));
  v = fmaxf(v, __shfl_xor(v, 4, 32));
  v = fmaxf(v, __shfl_xor(v, 8, 32));
  return v;
}
__device__ __forceinline__ float half16_sum(float v) {
  v += __shfl_xor(v, 1, 32);
  v += __shfl_xor(v, 2, 32);
  v += __shfl_xor(v, 4, 32);
  v += __shfl_xor(v, 8, 32);
  return v;
}

// ---------------------------------------------------------------------------
// Tensor Data Mover: DMA one 128-row x 32-f16 tile (row stride ldK elements)
// from global memory into LDS. D# per CDNA5 ISA ch.8 (2D mode, groups 2/3 zero).
// ---------------------------------------------------------------------------
__device__ __forceinline__ void tdm_load_tile_128x32(const _Float16* gbase, int ldK,
                                                     _Float16* ldsDst) {
  unsigned lds_addr = (unsigned)(uintptr_t)ldsDst;            // low 32b of flat shared addr = LDS offset
  unsigned long long ga = (unsigned long long)(uintptr_t)gbase;
  u32x4 g0 = {
    1u,                                                       // count=1, user descriptor
    lds_addr,                                                 // lds_addr
    (unsigned)(ga & 0xFFFFFFFFu),                             // global_addr[31:0]
    (unsigned)((ga >> 32) & 0x1FFFFFFu) | (2u << 30)          // global_addr[56:32] | type=2
  };
  i32x8 g1 = {
    1 << 16,                                                  // data_size=1 (2B), wg_mask=0, no pad/iter
    32 << 16,                                                 // tensor_dim0[15:0]=32 in bits[31:16]
    128 << 16,                                                // tensor_dim0 hi=0 | tensor_dim1[15:0]=128
    32 << 16,                                                 // tensor_dim1 hi=0 | tile_dim0=32
    128,                                                      // tile_dim1=128 | tile_dim2=0
    ldK,                                                      // tensor_dim0_stride[31:0]
    0,                                                        // stride hi | tensor_dim1_stride lo (unused, 2D)
    0
  };
  i32x4 z4 = {0, 0, 0, 0};
#if __clang_major__ >= 23
  i32x8 z8 = {0, 0, 0, 0, 0, 0, 0, 0};
  __builtin_amdgcn_tensor_load_to_lds(g0, g1, z4, z4, z8, 0);
#else
  __builtin_amdgcn_tensor_load_to_lds(g0, g1, z4, z4, 0);
#endif
}

// ---------------------------------------------------------------------------
// Elementwise converts / repacks
// ---------------------------------------------------------------------------
__global__ void f32_to_f16_kernel(const float* __restrict__ in, _Float16* __restrict__ out, int n) {
  int i = blockIdx.x * blockDim.x + threadIdx.x;
  if (i < n) out[i] = (_Float16)in[i];
}

// w: (K rows x N cols) f32 row-major -> wT: (N x K) f16 row-major
__global__ void transpose_f32_f16_kernel(const float* __restrict__ w, _Float16* __restrict__ wT,
                                         int K, int N) {
  int i = blockIdx.x * blockDim.x + threadIdx.x;
  if (i >= K * N) return;
  int k = i / N, n = i - k * N;
  wT[(size_t)n * K + k] = (_Float16)w[i];
}

// x: [S][nh*128] f32 -> RoPE -> out: [nh][S][128] f16. positions == arange(S).
__global__ void rope_pack_kernel(const float* __restrict__ x,
                                 const float* __restrict__ cosT,
                                 const float* __restrict__ sinT,
                                 _Float16* __restrict__ out, int nh) {
  int i = blockIdx.x * blockDim.x + threadIdx.x;
  int total = S_LEN * nh * 64;
  if (i >= total) return;
  int d = i & 63;
  int h = (i >> 6) % nh;
  int s = i / (64 * nh);
  float c  = cosT[s * 64 + d];
  float sn = sinT[s * 64 + d];
  const float* row = x + (size_t)s * nh * DHEAD + h * DHEAD;
  float x1 = row[d], x2 = row[d + 64];
  _Float16* orow = out + ((size_t)h * S_LEN + s) * DHEAD;
  orow[d]      = (_Float16)(x1 * c - x2 * sn);
  orow[d + 64] = (_Float16)(x2 * c + x1 * sn);
}

// v: [S][4*128] f32 -> vT: [4][128][S] f16 (so P*V B-fragments are row-contiguous)
__global__ void v_pack_kernel(const float* __restrict__ v, _Float16* __restrict__ vT) {
  int i = blockIdx.x * blockDim.x + threadIdx.x;
  if (i >= S_LEN * KVDIM) return;
  int c = i & (KVDIM - 1);
  int s = i >> 9;
  int kv = c >> 7, d = c & 127;
  vT[((size_t)kv * DHEAD + d) * S_LEN + s] = (_Float16)v[i];
}

// ---------------------------------------------------------------------------
// TDM-staged WMMA GEMM: C(MxN, f32) = A(MxK f16) * Bt^T, Bt is (NxK f16).
// Block = 256 threads = 8 waves; block tile 128x128; wave tile 32x64.
// Wave 0 drives the Tensor Data Mover (double-buffered 128x32 tiles in LDS,
// TENSORcnt-tracked); all 8 waves consume fragments via ds_load_b128.
// ---------------------------------------------------------------------------
__global__ __launch_bounds__(256) void gemm_f16_f32_kernel(const _Float16* __restrict__ A,
                                                           const _Float16* __restrict__ Bt,
                                                           float* __restrict__ C,
                                                           int M, int N, int K) {
  __shared__ _Float16 Atile[2][128 * 32];   // 8 KB each
  __shared__ _Float16 Btile[2][128 * 32];

  const int lane = threadIdx.x & 31;
  const int wid  = threadIdx.x >> 5;
  const int mBlock = blockIdx.y * 128;
  const int nBlock = blockIdx.x * 128;
  const int wm = (wid & 3) * 32;            // wave row offset in block tile
  const int wn = (wid >> 2) * 64;           // wave col offset in block tile

  if (wid == 0) {                           // uniform branch: EXEC stays all-ones per wave
    tdm_load_tile_128x32(A  + (size_t)mBlock * K, K, Atile[0]);
    tdm_load_tile_128x32(Bt + (size_t)nBlock * K, K, Btile[0]);
  }

  v8f acc[2][4] = {};
  int buf = 0;
  for (int kk = 0; kk < K; kk += 32, buf ^= 1) {
    const bool more = (kk + 32) < K;
    if (wid == 0) {
      if (more) {
        tdm_load_tile_128x32(A  + (size_t)mBlock * K + kk + 32, K, Atile[buf ^ 1]);
        tdm_load_tile_128x32(Bt + (size_t)nBlock * K + kk + 32, K, Btile[buf ^ 1]);
        __builtin_amdgcn_s_wait_tensorcnt(2);   // retire current pair, leave next pair in flight
      } else {
        __builtin_amdgcn_s_wait_tensorcnt(0);
      }
    }
    __syncthreads();                            // current buffer visible to all waves

    v16h a0 = load_frag16(Atile[buf], 32, wm,      0);
    v16h a1 = load_frag16(Atile[buf], 32, wm + 16, 0);
    v16h b0 = load_frag16(Btile[buf], 32, wn,      0);
    v16h b1 = load_frag16(Btile[buf], 32, wn + 16, 0);
    v16h b2 = load_frag16(Btile[buf], 32, wn + 32, 0);
    v16h b3 = load_frag16(Btile[buf], 32, wn + 48, 0);
    acc[0][0] = wmma_f16(a0, b0, acc[0][0]);
    acc[0][1] = wmma_f16(a0, b1, acc[0][1]);
    acc[0][2] = wmma_f16(a0, b2, acc[0][2]);
    acc[0][3] = wmma_f16(a0, b3, acc[0][3]);
    acc[1][0] = wmma_f16(a1, b0, acc[1][0]);
    acc[1][1] = wmma_f16(a1, b1, acc[1][1]);
    acc[1][2] = wmma_f16(a1, b2, acc[1][2]);
    acc[1][3] = wmma_f16(a1, b3, acc[1][3]);

    __syncthreads();                            // everyone done with `buf` before it is re-filled
  }

  const int col  = lane & 15;
  const int rofs = (lane >> 4) << 3;   // C layout: VGPR r -> row r (lanes 0-15) / r+8 (lanes 16-31)
#pragma unroll
  for (int i = 0; i < 2; ++i)
#pragma unroll
    for (int j = 0; j < 4; ++j)
#pragma unroll
      for (int r = 0; r < 8; ++r)
        C[(size_t)(mBlock + wm + i * 16 + r + rofs) * N + nBlock + wn + j * 16 + col] = acc[i][j][r];
}

// ---------------------------------------------------------------------------
// Flash attention: one wave per (head, 16-row Q block). Causal, GQA groups=4.
// qH: [16][S][128] f16, kH: [4][S][128] f16, vT: [4][128][S] f16,
// outH: [S][2048] f16 (row-major, head-interleaved, feeds the wo GEMM).
// ---------------------------------------------------------------------------
__global__ __launch_bounds__(256) void attn_kernel(const _Float16* __restrict__ qH,
                                                   const _Float16* __restrict__ kH,
                                                   const _Float16* __restrict__ vT,
                                                   _Float16* __restrict__ outH) {
  __shared__ _Float16 pbuf[8][16 * 32];

  const int lane = threadIdx.x & 31;
  const int wid  = threadIdx.x >> 5;
  const int job  = blockIdx.x * 8 + wid;       // 0 .. 4095
  const int head = job >> 8;                   // / (S/16)
  const int qb   = job & 255;
  const int kvh  = head >> 2;                  // groups = NH/NKV = 4
  const float scale = 0.08838834764831845f;    // 1/sqrt(128)

  const _Float16* qhead = qH + (size_t)head * S_LEN * DHEAD;
  const _Float16* khead = kH + (size_t)kvh  * S_LEN * DHEAD;
  const _Float16* vhead = vT + (size_t)kvh  * DHEAD * S_LEN;

  // Q fragments for the 4 d-chunks, resident in VGPRs for the whole key loop.
  v16h aq[4];
#pragma unroll
  for (int c = 0; c < 4; ++c) aq[c] = load_frag16(qhead, DHEAD, qb * 16, c * 32);

  float rmax[8], rsum[8];
  v8f o[8] = {};
#pragma unroll
  for (int r = 0; r < 8; ++r) { rmax[r] = -1e30f; rsum[r] = 0.0f; }

  const int col  = lane & 15;
  const int rofs = (lane >> 4) << 3;
  const int kend = qb * 16 + 16;               // causal horizon for this block

  for (int kk = 0; kk < kend; kk += 32) {
    // --- scores: two 16x16 tiles over a 32-key chunk, fp32 accum over D ---
    v8f sc[2] = {};
#pragma unroll
    for (int t = 0; t < 2; ++t)
#pragma unroll
      for (int c = 0; c < 4; ++c) {
        v16h bk = load_frag16(khead, DHEAD, kk + t * 16, c * 32);
        sc[t] = wmma_f16(aq[c], bk, sc[t]);
      }

    // --- online softmax update (row stats live in the 16-lane half owning the row) ---
    float pv0[8], pv1[8];
#pragma unroll
    for (int r = 0; r < 8; ++r) {
      int m = qb * 16 + r + rofs;
      float v0 = sc[0][r] * scale + ((kk + col)      <= m ? 0.0f : -1e9f);
      float v1 = sc[1][r] * scale + ((kk + 16 + col) <= m ? 0.0f : -1e9f);
      float nmax = fmaxf(rmax[r], half16_max(fmaxf(v0, v1)));
      float alpha = __expf(rmax[r] - nmax);
      rmax[r] = nmax;
      float p0 = __expf(v0 - nmax);
      float p1 = __expf(v1 - nmax);
      pv0[r] = p0; pv1[r] = p1;
      rsum[r] = rsum[r] * alpha + half16_sum(p0 + p1);
#pragma unroll
      for (int j = 0; j < 8; ++j) o[j][r] *= alpha;
    }

    // --- transpose P (C-layout -> A-layout) through per-wave LDS ---
    _Float16* pb = pbuf[wid];
#pragma unroll
    for (int r = 0; r < 8; ++r) {
      int mloc = r + rofs;
      pb[mloc * 32 + col]      = (_Float16)pv0[r];
      pb[mloc * 32 + 16 + col] = (_Float16)pv1[r];
    }
    asm volatile("s_wait_dscnt 0x0" ::: "memory");
    v16h ap = load_frag16(pb, 32, 0, 0);

    // --- o += P * V over the 32-key chunk, 8 d-tiles ---
#pragma unroll
    for (int j = 0; j < 8; ++j) {
      v16h bv = load_frag16(vhead, S_LEN, j * 16, kk);
      o[j] = wmma_f16(ap, bv, o[j]);
    }
  }

  // --- normalize and emit f16 rows for the output projection ---
  _Float16* obase = outH + (size_t)(qb * 16) * HDIM + head * DHEAD;
#pragma unroll
  for (int r = 0; r < 8; ++r) {
    float inv = 1.0f / rsum[r];
#pragma unroll
    for (int j = 0; j < 8; ++j)
      obase[(size_t)(r + rofs) * HDIM + j * 16 + col] = (_Float16)(o[j][r] * inv);
  }
}

// ---------------------------------------------------------------------------
// Host-side orchestration
// ---------------------------------------------------------------------------
extern "C" void kernel_launch(void* const* d_in, const int* in_sizes, int n_in,
                              void* d_out, int out_size, void* d_ws, size_t ws_size,
                              hipStream_t stream) {
  const float* hs   = (const float*)d_in[0];   // (1, 4096, 2048)
  const float* cosT = (const float*)d_in[1];   // (4096, 64)
  const float* sinT = (const float*)d_in[2];   // (4096, 64)
  const float* wq   = (const float*)d_in[8];   // (2048, 2048)
  const float* wk   = (const float*)d_in[9];   // (2048, 512)
  const float* wv   = (const float*)d_in[10];  // (2048, 512)
  const float* wo   = (const float*)d_in[11];  // (2048, 2048)
  float* out = (float*)d_out;                  // (1, 4096, 2048) f32

  char* ws = (char*)d_ws;
  size_t off = 0;
  _Float16* hsH  = (_Float16*)(ws + off); off += (size_t)S_LEN * HDIM * 2;       // 16 MB
  _Float16* wqT  = (_Float16*)(ws + off); off += (size_t)HDIM * HDIM * 2;        //  8 MB
  _Float16* wkT  = (_Float16*)(ws + off); off += (size_t)KVDIM * HDIM * 2;       //  2 MB
  _Float16* wvT  = (_Float16*)(ws + off); off += (size_t)KVDIM * HDIM * 2;       //  2 MB
  _Float16* woT  = (_Float16*)(ws + off); off += (size_t)HDIM * HDIM * 2;        //  8 MB
  float*    qF   = (float*)(ws + off);    off += (size_t)S_LEN * HDIM * 4;       // 32 MB
  float*    kF   = (float*)(ws + off);    off += (size_t)S_LEN * KVDIM * 4;      //  8 MB
  float*    vF   = (float*)(ws + off);    off += (size_t)S_LEN * KVDIM * 4;      //  8 MB
  _Float16* qHh  = (_Float16*)(ws + off); off += (size_t)NHQ * S_LEN * DHEAD * 2;// 16 MB
  _Float16* kHh  = (_Float16*)(ws + off); off += (size_t)NKV * S_LEN * DHEAD * 2;//  4 MB
  _Float16* vTh  = (_Float16*)(ws + off); off += (size_t)NKV * DHEAD * S_LEN * 2;//  4 MB
  _Float16* attnH= (_Float16*)(ws + off); off += (size_t)S_LEN * HDIM * 2;       // 16 MB

  // 1) convert activations / weights to f16 (weights transposed to NxK)
  {
    int n = S_LEN * HDIM;
    f32_to_f16_kernel<<<(n + 255) / 256, 256, 0, stream>>>(hs, hsH, n);
  }
  transpose_f32_f16_kernel<<<(HDIM * HDIM + 255) / 256, 256, 0, stream>>>(wq, wqT, HDIM, HDIM);
  transpose_f32_f16_kernel<<<(HDIM * KVDIM + 255) / 256, 256, 0, stream>>>(wk, wkT, HDIM, KVDIM);
  transpose_f32_f16_kernel<<<(HDIM * KVDIM + 255) / 256, 256, 0, stream>>>(wv, wvT, HDIM, KVDIM);
  transpose_f32_f16_kernel<<<(HDIM * HDIM + 255) / 256, 256, 0, stream>>>(wo, woT, HDIM, HDIM);

  // 2) QKV projections (TDM-staged WMMA GEMMs, fp32 out)
  gemm_f16_f32_kernel<<<dim3(HDIM / 128, S_LEN / 128), 256, 0, stream>>>(hsH, wqT, qF, S_LEN, HDIM, HDIM);
  gemm_f16_f32_kernel<<<dim3(KVDIM / 128, S_LEN / 128), 256, 0, stream>>>(hsH, wkT, kF, S_LEN, KVDIM, HDIM);
  gemm_f16_f32_kernel<<<dim3(KVDIM / 128, S_LEN / 128), 256, 0, stream>>>(hsH, wvT, vF, S_LEN, KVDIM, HDIM);

  // 3) RoPE + repack to WMMA-friendly per-head layouts
  rope_pack_kernel<<<(S_LEN * NHQ * 64 + 255) / 256, 256, 0, stream>>>(qF, cosT, sinT, qHh, NHQ);
  rope_pack_kernel<<<(S_LEN * NKV * 64 + 255) / 256, 256, 0, stream>>>(kF, cosT, sinT, kHh, NKV);
  v_pack_kernel<<<(S_LEN * KVDIM + 255) / 256, 256, 0, stream>>>(vF, vTh);

  // 4) causal GQA flash attention (one wave per head x 16-row Q block)
  attn_kernel<<<(NHQ * (S_LEN / 16)) / 8, 256, 0, stream>>>(qHh, kHh, vTh, attnH);

  // 5) output projection -> fp32 result
  gemm_f16_f32_kernel<<<dim3(HDIM / 128, S_LEN / 128), 256, 0, stream>>>(attnH, woT, out, S_LEN, HDIM, HDIM);
}
